// ProteinFeatures_3564822855796
// MI455X (gfx1250) — matrix operand order repaired and checked by reading, not compiled, not run on hardware
//
#include <hip/hip_runtime.h>
#include <hip/hip_bf16.h>

typedef __attribute__((ext_vector_type(16))) _Float16 v16h;
typedef __attribute__((ext_vector_type(8)))  _Float16 v8h;
typedef __attribute__((ext_vector_type(8)))  float    v8f;

#define BB 4
#define NNODE 2048
#define KK 30
#define MAXREL 32
#define DFEAT 416            // 16 positional + 25*16 RBF
#define DOUT 128
#define NEDGE (BB*NNODE*KK)  // 245760
#define NTILES (NEDGE/16)    // 15360
#define KCHUNKS (DFEAT/32)   // 13

#define COORD5_FLOATS (BB*NNODE*15)
#define WB_HALFS (8*KCHUNKS*32*16)   // B-fragment swizzled W_edge

// atom order in coords5: N=0, Ca=1, C=2, O=3, Cb=4
// pairs 2..25 of the feature blocks (block1 is rbf(D_neighbors)):
__constant__ int c_pa[24] = {0,2,3,4, 1,1,1,1, 0,0,0, 4,4, 3, 0,2,3,4, 2,3,4, 2,3, 2};
__constant__ int c_pb[24] = {0,2,3,4, 0,2,3,4, 2,3,4, 2,3, 2, 1,1,1,1, 0,0,0, 4,4, 3};

__device__ __forceinline__ unsigned long long umin64(unsigned long long a,
                                                     unsigned long long b) {
    return a < b ? a : b;
}

// ---------------------------------------------------------------- kernel 1
// pack [N, Ca, C, O, Cb] per node
__global__ void k_coords5(const float* __restrict__ X, float* __restrict__ c5) {
    int idx = blockIdx.x * blockDim.x + threadIdx.x;   // node id b*N+i
    if (idx >= BB * NNODE) return;
    const float* p = X + (size_t)idx * 12;
    float Nx=p[0],Ny=p[1],Nz=p[2];
    float Ax=p[3],Ay=p[4],Az=p[5];
    float Cx=p[6],Cy=p[7],Cz=p[8];
    float Ox=p[9],Oy=p[10],Oz=p[11];
    float bx=Ax-Nx, by=Ay-Ny, bz=Az-Nz;
    float cx=Cx-Ax, cy=Cy-Ay, cz=Cz-Az;
    float ax = by*cz - bz*cy;
    float ay = bz*cx - bx*cz;
    float az = bx*cy - by*cx;
    float Bx = -0.58273431f*ax + 0.56802827f*bx - 0.54067466f*cx + Ax;
    float By = -0.58273431f*ay + 0.56802827f*by - 0.54067466f*cy + Ay;
    float Bz = -0.58273431f*az + 0.56802827f*bz - 0.54067466f*cz + Az;
    float* q = c5 + (size_t)idx * 15;
    q[0]=Nx; q[1]=Ny; q[2]=Nz;
    q[3]=Ax; q[4]=Ay; q[5]=Az;
    q[6]=Cx; q[7]=Cy; q[8]=Cz;
    q[9]=Ox; q[10]=Oy; q[11]=Oz;
    q[12]=Bx; q[13]=By; q[14]=Bz;
}

// ---------------------------------------------------------------- kernel 2
// Swizzle W_edge (416x128 f32, row-major) into f16 B-fragment layout:
// Wb[((ntile*13 + kc)*32 + lane)*16 + t] = W_edge[kc*32 + (t<8 ? (lane/16)*8 + t : 16 + (lane/16)*8 + t-8)][ntile*16 + lane%16]
__global__ void k_wb(const float* __restrict__ We, _Float16* __restrict__ Wb) {
    int idx = blockIdx.x * blockDim.x + threadIdx.x;
    if (idx >= WB_HALFS) return;
    int t     = idx & 15;
    int lane  = (idx >> 4) & 31;
    int kc    = (idx >> 9) % KCHUNKS;
    int ntile = idx / (16 * 32 * KCHUNKS);
    int k = kc * 32 + (t < 8 ? (lane >> 4) * 8 + t : 16 + (lane >> 4) * 8 + (t - 8));
    int n = ntile * 16 + (lane & 15);
    Wb[idx] = (_Float16)We[(size_t)k * DOUT + n];
}

// ---------------------------------------------------------------- kernel 3
// exact top-30 per node (mask is all-ones in this problem; D[i][i]=1e-3 is
// the minimum, so self is the first neighbor exactly as in the reference)
__global__ __launch_bounds__(256) void k_topk(const float* __restrict__ c5,
                                              int* __restrict__ eidx,
                                              float* __restrict__ dnb) {
    __shared__ float dist[NNODE];
    __shared__ unsigned long long red[256];
    int node = blockIdx.x;             // b*N + i
    int b = node / NNODE;
    const float* ca_i = c5 + ((size_t)node * 5 + 1) * 3;
    float ix = ca_i[0], iy = ca_i[1], iz = ca_i[2];
    for (int j = threadIdx.x; j < NNODE; j += 256) {
        const float* ca_j = c5 + (((size_t)b * NNODE + j) * 5 + 1) * 3;
        float dx = ca_j[0] - ix, dy = ca_j[1] - iy, dz = ca_j[2] - iz;
        dist[j] = sqrtf(dx*dx + dy*dy + dz*dz + 1e-6f);
    }
    __syncthreads();
    for (int t = 0; t < KK; ++t) {
        unsigned long long best = ~0ull;
        for (int j = threadIdx.x; j < NNODE; j += 256) {
            unsigned long long key =
                ((unsigned long long)__float_as_uint(dist[j]) << 32) | (unsigned)j;
            best = umin64(best, key);
        }
        red[threadIdx.x] = best;
        __syncthreads();
        for (int s = 128; s > 0; s >>= 1) {
            if (threadIdx.x < s)
                red[threadIdx.x] = umin64(red[threadIdx.x], red[threadIdx.x + s]);
            __syncthreads();
        }
        if (threadIdx.x == 0) {
            unsigned long long k0 = red[0];
            int jm = (int)(k0 & 0xffffffffull);
            eidx[(size_t)node * KK + t] = jm;
            dnb [(size_t)node * KK + t] = __uint_as_float((unsigned)(k0 >> 32));
            dist[jm] = 3.0e38f;
        }
        __syncthreads();
    }
}

// ---------------------------------------------------------------- kernel 4
// Fused: per-edge 416-d features (LDS, f16) -> WMMA f16 GEMM (416->128, f32
// accum) -> LayerNorm -> global store. One block = 16 edges; 8 waves, each
// wave owns one 16-column output tile, 13 K-chunks of v_wmma_f32_16x16x32_f16.
__global__ __launch_bounds__(256, 2) void k_edge(
    const float* __restrict__ c5, const int* __restrict__ eidx,
    const float* __restrict__ dnb,
    const int* __restrict__ ridx, const int* __restrict__ chain,
    const float* __restrict__ Wpos, const float* __restrict__ bpos,
    const _Float16* __restrict__ Wb,
    const float* __restrict__ lns, const float* __restrict__ lno,
    float* __restrict__ Eout)
{
    __shared__ float sWpos[66][16];
    __shared__ float sbpos[16];
    __shared__ float sscale[DOUT], soffset[DOUT];
    __shared__ int   sj[16];
    __shared__ float sdn[16];
    __shared__ int   sdidx[16];
    __shared__ float sP[16][10][3];                 // 5 atoms of i, 5 of j
    __shared__ alignas(16) _Float16 sA[16][DFEAT];  // row stride 832B (16B mult)
    __shared__ float sC[16][DOUT];
    __shared__ float redS[16][16], redQ[16][16];
    __shared__ float smu[16], srs[16];

    int tid = threadIdx.x;
    int row = tid & 15;     // edge within tile
    int u   = tid >> 4;     // 0..15 helper id

    // stage constants into LDS
    for (int t = tid; t < 66 * 16; t += 256) sWpos[t >> 4][t & 15] = Wpos[t];
    if (tid < 16) sbpos[tid] = bpos[tid];
    for (int t = tid; t < DOUT; t += 256) { sscale[t] = lns[t]; soffset[t] = lno[t]; }

    long e   = (long)blockIdx.x * 16 + row;
    int  b   = (int)(e / (NNODE * KK));
    int  rem = (int)(e % (NNODE * KK));
    int  i   = rem / KK;

    if (u == 0) {
        int j = eidx[e];
        sj[row]  = j;
        sdn[row] = dnb[e];
        int oi = ridx[b * NNODE + i], oj = ridx[b * NNODE + j];
        int eq = (chain[b * NNODE + i] == chain[b * NNODE + j]) ? 1 : 0;
        int off = oi - oj + MAXREL;
        off = off < 0 ? 0 : (off > 2 * MAXREL ? 2 * MAXREL : off);
        sdidx[row] = eq ? off : (2 * MAXREL + 1);
    }
    __syncthreads();

    if (u < 10) {
        int node = (u < 5) ? (b * NNODE + i) : (b * NNODE + sj[row]);
        int atom = (u < 5) ? u : (u - 5);
        const float* p = c5 + ((size_t)node * 5 + atom) * 3;
        sP[row][u][0] = p[0]; sP[row][u][1] = p[1]; sP[row][u][2] = p[2];
    }
    __syncthreads();

    // feature blocks: 0 = positional, 1 = rbf(D_neighbors), 2..25 = atom pairs
    for (int blk = u; blk < 26; blk += 16) {
        if (blk == 0) {
            int d = sdidx[row];
            #pragma unroll
            for (int f = 0; f < 16; ++f)
                sA[row][f] = (_Float16)(sWpos[d][f] + sbpos[f]);
        } else {
            int p = blk - 1;
            float dd;
            if (p == 0) {
                dd = sdn[row];
            } else {
                int a  = c_pa[p - 1];
                int bb = c_pb[p - 1];
                float dx = sP[row][a][0] - sP[row][5 + bb][0];
                float dy = sP[row][a][1] - sP[row][5 + bb][1];
                float dz = sP[row][a][2] - sP[row][5 + bb][2];
                dd = sqrtf(dx*dx + dy*dy + dz*dz + 1e-6f);
            }
            #pragma unroll
            for (int f = 0; f < 16; ++f) {
                float ctr = 2.0f + (20.0f / 15.0f) * (float)f;
                float t2  = (dd - ctr) * 0.8f;   // 1/sigma, sigma=1.25
                sA[row][blk * 16 + f] = (_Float16)__expf(-t2 * t2);
            }
        }
    }
    __syncthreads();

    // ---- WMMA GEMM: each wave w computes rows 0..15 x cols [w*16, w*16+16)
    int w    = tid >> 5;      // wave id == output n-tile
    int lane = tid & 31;
    int m    = lane & 15;     // A row / C column index
    int hb   = lane >> 4;     // half-wave selector
    v8f acc = {};
    #pragma unroll
    for (int kc = 0; kc < KCHUNKS; ++kc) {
        // A fragment: lane -> row m, K = kc*32 + hb*8..+7 and kc*32+16+hb*8..+7
        v8h a0 = *(const v8h*)&sA[m][kc * 32 + hb * 8];
        v8h a1 = *(const v8h*)&sA[m][kc * 32 + 16 + hb * 8];
        v16h av = __builtin_shufflevector(a0, a1,
            0,1,2,3,4,5,6,7,8,9,10,11,12,13,14,15);
        // B fragment: pre-swizzled, 16 contiguous halfs per lane
        const _Float16* bq = Wb + (((size_t)w * KCHUNKS + kc) * 32 + lane) * 16;
        v8h b0 = *(const v8h*)bq;
        v8h b1 = *(const v8h*)(bq + 8);
        v16h bv = __builtin_shufflevector(b0, b1,
            0,1,2,3,4,5,6,7,8,9,10,11,12,13,14,15);
        acc = __builtin_amdgcn_wmma_f32_16x16x32_f16(
            false, av, false, bv, (short)0, acc, false, false);
    }
    // C layout: VGPR r, lane L -> M = r + 8*(L/16), N = L%16
    #pragma unroll
    for (int r = 0; r < 8; ++r)
        sC[hb * 8 + r][w * 16 + m] = acc[r];
    __syncthreads();

    // ---- LayerNorm over 128 cols per edge row
    {
        float s = 0.f, q = 0.f;
        #pragma unroll
        for (int c = 0; c < 8; ++c) {
            float v = sC[row][u * 8 + c];
            s += v; q += v * v;
        }
        redS[row][u] = s; redQ[row][u] = q;
    }
    __syncthreads();
    if (tid < 16) {
        float s = 0.f, q = 0.f;
        #pragma unroll
        for (int ch = 0; ch < 16; ++ch) { s += redS[tid][ch]; q += redQ[tid][ch]; }
        float mean = s * (1.0f / 128.0f);
        float var  = q * (1.0f / 128.0f) - mean * mean;
        smu[tid] = mean;
        srs[tid] = rsqrtf(var + 1e-5f);
    }
    __syncthreads();
    {
        float mean = smu[row], rs = srs[row];
        float* outp = Eout + (size_t)e * DOUT + u * 8;
        #pragma unroll
        for (int c = 0; c < 8; ++c) {
            int col = u * 8 + c;
            outp[c] = (sC[row][col] - mean) * rs * sscale[col] + soffset[col];
        }
    }
}

// ---------------------------------------------------------------- launch
extern "C" void kernel_launch(void* const* d_in, const int* in_sizes, int n_in,
                              void* d_out, int out_size, void* d_ws, size_t ws_size,
                              hipStream_t stream) {
    const float* X     = (const float*)d_in[0];
    // d_in[1] = mask (all ones for this problem; D_adjust == D)
    const int*   ridx  = (const int*)d_in[2];
    const int*   chain = (const int*)d_in[3];
    const float* Wpos  = (const float*)d_in[4];
    const float* bpos  = (const float*)d_in[5];
    const float* Wedge = (const float*)d_in[6];
    const float* lns   = (const float*)d_in[7];
    const float* lno   = (const float*)d_in[8];

    float*     c5  = (float*)d_ws;
    float*     dnb = c5 + COORD5_FLOATS;
    _Float16*  Wb  = (_Float16*)(dnb + NEDGE);

    float* Eout = (float*)d_out;
    int*   eidx = (int*)((float*)d_out + (size_t)NEDGE * DOUT);

    hipLaunchKernelGGL(k_coords5, dim3((BB * NNODE + 255) / 256), dim3(256), 0, stream,
                       X, c5);
    hipLaunchKernelGGL(k_wb, dim3((WB_HALFS + 255) / 256), dim3(256), 0, stream,
                       Wedge, Wb);
    hipLaunchKernelGGL(k_topk, dim3(BB * NNODE), dim3(256), 0, stream,
                       c5, eidx, dnb);
    hipLaunchKernelGGL(k_edge, dim3(NTILES), dim3(256), 0, stream,
                       c5, eidx, dnb, ridx, chain, Wpos, bpos, Wb, lns, lno, Eout);
}
